// Model_GRU_pip_10642928959816
// MI455X (gfx1250) — matrix operand order repaired and checked by reading, not compiled, not run on hardware
//
#include <hip/hip_runtime.h>
#include <hip/hip_bf16.h>
#include <cstdint>

// ---------------- problem constants ----------------
#define NROWS      262144          // 4 * 65536
#define ROW        386             // 2 + 256 + 128
#define DIM_MAIL   256
#define DIM_MEM    128
#define DIM_TIME   128
#define GRU_IN     384             // DIM_MAIL + DIM_TIME
#define NW_IH      (3 * DIM_MEM * GRU_IN)    // 147456
#define NW_HH      (3 * DIM_MEM * DIM_MEM)   // 49152
#define MT         64              // rows per block

typedef __bf16 bf16;
typedef __attribute__((ext_vector_type(8)))  __bf16 v8bf;
typedef __attribute__((ext_vector_type(16))) __bf16 v16bf;
typedef __attribute__((ext_vector_type(8)))  float  v8f;
typedef __attribute__((ext_vector_type(4)))  unsigned int u32x4;
typedef __attribute__((ext_vector_type(4)))  int i32x4;
typedef __attribute__((ext_vector_type(8)))  int i32x8;

union ABf  { v16bf v; v8bf h[2]; };
union PK2  { bf16 b[2]; unsigned u; };

// ---------------- weight conversion: f32 -> bf16 panels in workspace ----------------
__global__ __launch_bounds__(256) void cvt_weights_kernel(const float* __restrict__ Wih,
                                                          const float* __restrict__ Whh,
                                                          bf16* __restrict__ out) {
    int i = blockIdx.x * 256 + threadIdx.x;
    if (i < NW_IH) out[i] = (bf16)Wih[i];
    if (i < NW_HH) out[NW_IH + i] = (bf16)Whh[i];
}

// ---------------- fused GRU update ----------------
__global__ __launch_bounds__(256) void gru_wmma_kernel(
    const float* __restrict__ data, const float* __restrict__ data_ts,
    const float* __restrict__ time_w, const float* __restrict__ time_b,
    const bf16* __restrict__ Wih_bf, const float* __restrict__ b_ih,
    const bf16* __restrict__ Whh_bf, const float* __restrict__ b_hh,
    float* __restrict__ out)
{
    __shared__ __align__(16) bf16  sX [MT][GRU_IN];   // [mail | time_feat] bf16 (48 KiB)
    __shared__ __align__(16) bf16  sH [MT][DIM_MEM];  // memory state bf16     (16 KiB)
    __shared__ __align__(16) float sHf[MT][DIM_MEM];  // memory state f32      (32 KiB)

    const int tid  = threadIdx.x;
    const int row0 = blockIdx.x * MT;

    // Warm the shared bf16 weight panels into cache (hot across all 4096 blocks).
    __builtin_prefetch(Wih_bf + (tid << 6), 0, 1);
    __builtin_prefetch(Whh_bf + (tid << 5), 0, 1);

    // ---------- stage 0a: DMA the f32 memory-state tile straight into LDS ----------
    // 2D tile: 64 rows x 128 f32, global row stride 386 elements -> one TDM descriptor.
#if __has_builtin(__builtin_amdgcn_tensor_load_to_lds) && __has_builtin(__builtin_amdgcn_s_wait_tensorcnt)
    if (tid < 32) {   // single wave issues the DMA (TDM ignores EXEC; avoid x8 duplicates)
        const unsigned long long ga =
            (unsigned long long)(uintptr_t)(data + (size_t)row0 * ROW + 2 + DIM_MAIL);
        u32x4 g0;
        g0[0] = 1u;                                          // count=1 (valid user D#)
        g0[1] = (unsigned)(uintptr_t)&sHf[0][0];             // lds_addr
        g0[2] = (unsigned)(ga & 0xFFFFFFFFu);                // global_addr[31:0]
        g0[3] = (unsigned)((ga >> 32) & 0x01FFFFFFu) | (2u << 30);  // addr[56:32] | type=2
        i32x8 g1;
        g1[0] = (int)(2u << 16);                             // data_size=2 (4 bytes)
        g1[1] = (int)((unsigned)DIM_MEM << 16);              // tensor_dim0 = 128 (lo16)
        g1[2] = (int)((unsigned)MT << 16);                   // tensor_dim1 = 64  (lo16)
        g1[3] = (int)((unsigned)DIM_MEM << 16);              // tile_dim0  = 128
        g1[4] = (int)MT;                                     // tile_dim1 = 64, tile_dim2 = 0
        g1[5] = ROW;                                         // tensor_dim0_stride = 386
        g1[6] = 0;
        g1[7] = 0;
        const i32x4 gz = {0, 0, 0, 0};                       // 2D tensor: groups 2/3 unused
#if __clang_major__ >= 23
        const i32x8 gz8 = {0, 0, 0, 0, 0, 0, 0, 0};
        __builtin_amdgcn_tensor_load_to_lds(g0, g1, gz, gz, gz8, 0);
#else
        __builtin_amdgcn_tensor_load_to_lds(g0, g1, gz, gz, 0);
#endif
    }
#else
    // Fallback: per-lane async copies (rows only 8B-aligned -> B64 flavor).
    #pragma unroll
    for (int i = 0; i < 16; ++i) {
        int q = tid + 256 * i;            // 4096 8-byte chunks (64 rows x 512B)
        int m = q >> 6, c2 = q & 63;
        unsigned lds_dst = (unsigned)(uintptr_t)&sHf[m][c2 * 2];
        const float* gsrc = &data[(size_t)(row0 + m) * ROW + 2 + DIM_MAIL + c2 * 2];
        asm volatile("global_load_async_to_lds_b64 %0, %1, off"
                     :: "v"(lds_dst), "v"(gsrc) : "memory");
    }
#endif

    // ---------- stage 0b: build bf16 X panel while the DMA runs ----------
    // mail inputs: 64 rows x 256 f32 -> bf16 (float2 chunks, 8B aligned)
    #pragma unroll
    for (int i = 0; i < 32; ++i) {
        int q = tid + 256 * i;            // 8192 float2 chunks
        int m = q >> 7, c2 = q & 127;
        float2 v = *reinterpret_cast<const float2*>(&data[(size_t)(row0 + m) * ROW + 2 + c2 * 2]);
        PK2 pk; pk.b[0] = (bf16)v.x; pk.b[1] = (bf16)v.y;
        *reinterpret_cast<unsigned*>(&sX[m][c2 * 2]) = pk.u;
    }
    // time features: cos((ts - mem_ts) * w + b), 64 rows x 64 pairs
    #pragma unroll
    for (int i = 0; i < 16; ++i) {
        int q = tid + 256 * i;            // 4096 pairs
        int m = q >> 6, j = (q & 63) * 2;
        float dt = data_ts[row0 + m] - data[(size_t)(row0 + m) * ROW];
        float f0 = __cosf(dt * time_w[j]     + time_b[j]);
        float f1 = __cosf(dt * time_w[j + 1] + time_b[j + 1]);
        PK2 pk; pk.b[0] = (bf16)f0; pk.b[1] = (bf16)f1;
        *reinterpret_cast<unsigned*>(&sX[m][DIM_MAIL + j]) = pk.u;
    }

    // Drain the DMA, then make it visible to the workgroup.
#if __has_builtin(__builtin_amdgcn_tensor_load_to_lds) && __has_builtin(__builtin_amdgcn_s_wait_tensorcnt)
    if (tid < 32) __builtin_amdgcn_s_wait_tensorcnt((short)0);
#else
    asm volatile("s_wait_asynccnt 0" ::: "memory");
#endif
    __syncthreads();

    // ---------- stage 0c: derive bf16 H panel from the DMA'd f32 copy ----------
    #pragma unroll
    for (int i = 0; i < 16; ++i) {
        int q = tid + 256 * i;            // 4096 float2 chunks
        int m = q >> 6, c2 = q & 63;
        float2 v = *reinterpret_cast<const float2*>(&sHf[m][c2 * 2]);
        PK2 pk; pk.b[0] = (bf16)v.x; pk.b[1] = (bf16)v.y;
        *reinterpret_cast<unsigned*>(&sH[m][c2 * 2]) = pk.u;
    }
    __syncthreads();

    // ---------- stage 1: WMMA GEMMs ----------
    // 8 waves: mi = wave&3 selects the 16-row M sub-tile (A-frag reused x4),
    //          g  = wave>>2 selects 4 of the 8 j-tiles: {g, g+2, g+4, g+6}.
    const int wave = tid >> 5, lane = tid & 31;
    const int mrow = (wave & 3) * 16;
    const int g    = wave >> 2;
    const int lrow = lane & 15;
    const int kh   = (lane < 16) ? 0 : 16;     // A-frag K-half (contiguous, matches B)
    const int khB  = (lane < 16) ? 0 : 16;     // B-frag K-half (contiguous, matches A)

    int jcol[4];
    #pragma unroll
    for (int p = 0; p < 4; ++p) jcol[p] = (g + 2 * p) * 16 + (lane & 15);

    const v8f vzero = {0.f, 0.f, 0.f, 0.f, 0.f, 0.f, 0.f, 0.f};
    v8f GIr[4], GIz[4], GIn[4], GHr[4], GHz[4], GHn[4];
    #pragma unroll
    for (int p = 0; p < 4; ++p) {
        GIr[p] = vzero; GIz[p] = vzero; GIn[p] = vzero;
        GHr[p] = vzero; GHz[p] = vzero; GHn[p] = vzero;
    }

    // gi = X @ Wih^T  (K = 384 -> 12 k-steps, 12 WMMA each)
    #pragma unroll
    for (int ks = 0; ks < 12; ++ks) {
        const int k0 = ks * 32;
        ABf a;
        a.h[0] = *reinterpret_cast<const v8bf*>(&sX[mrow + lrow][k0 + kh]);
        a.h[1] = *reinterpret_cast<const v8bf*>(&sX[mrow + lrow][k0 + kh + 8]);
        #pragma unroll
        for (int p = 0; p < 4; ++p) {
            ABf b;
            const v8bf* qr = reinterpret_cast<const v8bf*>(Wih_bf + (size_t)(jcol[p]) * GRU_IN + k0 + khB);
            b.h[0] = qr[0]; b.h[1] = qr[1];
            GIr[p] = __builtin_amdgcn_wmma_f32_16x16x32_bf16(false, a.v, false, b.v, (short)0, GIr[p], false, false);
            const v8bf* qz = reinterpret_cast<const v8bf*>(Wih_bf + (size_t)(128 + jcol[p]) * GRU_IN + k0 + khB);
            b.h[0] = qz[0]; b.h[1] = qz[1];
            GIz[p] = __builtin_amdgcn_wmma_f32_16x16x32_bf16(false, a.v, false, b.v, (short)0, GIz[p], false, false);
            const v8bf* qn = reinterpret_cast<const v8bf*>(Wih_bf + (size_t)(256 + jcol[p]) * GRU_IN + k0 + khB);
            b.h[0] = qn[0]; b.h[1] = qn[1];
            GIn[p] = __builtin_amdgcn_wmma_f32_16x16x32_bf16(false, a.v, false, b.v, (short)0, GIn[p], false, false);
        }
    }

    // gh = H @ Whh^T  (K = 128 -> 4 k-steps, 12 WMMA each)
    #pragma unroll
    for (int ks = 0; ks < 4; ++ks) {
        const int k0 = ks * 32;
        ABf a;
        a.h[0] = *reinterpret_cast<const v8bf*>(&sH[mrow + lrow][k0 + kh]);
        a.h[1] = *reinterpret_cast<const v8bf*>(&sH[mrow + lrow][k0 + kh + 8]);
        #pragma unroll
        for (int p = 0; p < 4; ++p) {
            ABf b;
            const v8bf* qr = reinterpret_cast<const v8bf*>(Whh_bf + (size_t)(jcol[p]) * DIM_MEM + k0 + khB);
            b.h[0] = qr[0]; b.h[1] = qr[1];
            GHr[p] = __builtin_amdgcn_wmma_f32_16x16x32_bf16(false, a.v, false, b.v, (short)0, GHr[p], false, false);
            const v8bf* qz = reinterpret_cast<const v8bf*>(Whh_bf + (size_t)(128 + jcol[p]) * DIM_MEM + k0 + khB);
            b.h[0] = qz[0]; b.h[1] = qz[1];
            GHz[p] = __builtin_amdgcn_wmma_f32_16x16x32_bf16(false, a.v, false, b.v, (short)0, GHz[p], false, false);
            const v8bf* qn = reinterpret_cast<const v8bf*>(Whh_bf + (size_t)(256 + jcol[p]) * DIM_MEM + k0 + khB);
            b.h[0] = qn[0]; b.h[1] = qn[1];
            GHn[p] = __builtin_amdgcn_wmma_f32_16x16x32_bf16(false, a.v, false, b.v, (short)0, GHn[p], false, false);
        }
    }

    // ---------- stage 2: gate epilogue entirely in registers ----------
    // C/D layout: VGPR r, lanes 0-15 -> (M=r, N=lane); lanes 16-31 -> (M=r+8, N=lane-16)
    #pragma unroll
    for (int p = 0; p < 4; ++p) {
        const int jc  = jcol[p];
        const float bi_r = b_ih[jc], bi_z = b_ih[128 + jc], bi_n = b_ih[256 + jc];
        const float bh_r = b_hh[jc], bh_z = b_hh[128 + jc], bh_n = b_hh[256 + jc];
        #pragma unroll
        for (int r = 0; r < 8; ++r) {
            const int m  = mrow + r + ((lane >= 16) ? 8 : 0);
            const float gr = GIr[p][r] + bi_r + GHr[p][r] + bh_r;
            const float gz = GIz[p][r] + bi_z + GHz[p][r] + bh_z;
            const float rr = 1.f / (1.f + __expf(-gr));
            const float zz = 1.f / (1.f + __expf(-gz));
            const float ga = GIn[p][r] + bi_n + rr * (GHn[p][r] + bh_n);
            const float e2 = __expf(2.f * ga);
            const float nn = (e2 - 1.f) / (e2 + 1.f);     // tanh
            const float h  = sHf[m][jc];
            out[(size_t)(row0 + m) * DIM_MEM + jc] = (1.f - zz) * nn + zz * h;
        }
    }
}

extern "C" void kernel_launch(void* const* d_in, const int* in_sizes, int n_in,
                              void* d_out, int out_size, void* d_ws, size_t ws_size,
                              hipStream_t stream) {
    const float* data    = (const float*)d_in[0];
    const float* data_ts = (const float*)d_in[1];
    const float* time_w  = (const float*)d_in[2];
    const float* time_b  = (const float*)d_in[3];
    const float* W_ih    = (const float*)d_in[4];
    const float* b_ih    = (const float*)d_in[5];
    const float* W_hh    = (const float*)d_in[6];
    const float* b_hh    = (const float*)d_in[7];
    float* out = (float*)d_out;
    bf16* wbf  = (bf16*)d_ws;   // [NW_IH bf16 Wih | NW_HH bf16 Whh] = 384 KiB

    cvt_weights_kernel<<<(NW_IH + 255) / 256, 256, 0, stream>>>(W_ih, W_hh, wbf);
    gru_wmma_kernel<<<NROWS / MT, 256, 0, stream>>>(
        data, data_ts, time_w, time_b, wbf, b_ih, wbf + NW_IH, b_hh, out);
}
